// MultiHeadAttention_49486613185208
// MI455X (gfx1250) — matrix operand order repaired and checked by reading, not compiled, not run on hardware
//
#include <hip/hip_runtime.h>

typedef __attribute__((ext_vector_type(16))) _Float16 v16h;
typedef __attribute__((ext_vector_type(8)))  float    v8f;
typedef __attribute__((ext_vector_type(4)))  _Float16 v4h;

#define NB 2            // batches per block
#define TOK_PAD 136     // token row stride in halfs (272B -> clean LDS banking)
#define D 128

union AFrag { v16h v; uint4 u[2]; };

// ---- DPP 16-lane row reduction (row_ror rotations, pure VALU) ----
template<int CTRL>
__device__ __forceinline__ float dpp_addf(float x) {
  return x + __int_as_float(__builtin_amdgcn_update_dpp(
      0, __float_as_int(x), CTRL, 0xF, 0xF, true));
}
__device__ __forceinline__ float rowsum16(float x) {
  x = dpp_addf<0x128>(x);   // row_ror:8
  x = dpp_addf<0x124>(x);   // row_ror:4
  x = dpp_addf<0x122>(x);   // row_ror:2
  x = dpp_addf<0x121>(x);   // row_ror:1
  return x;
}

__device__ __forceinline__ v16h load_bfrag(const _Float16* base) {
  AFrag f;
  const uint4* p = (const uint4*)base;
  f.u[0] = p[0]; f.u[1] = p[1];
  return f.v;
}
__device__ __forceinline__ v16h load_afrag_lds(const _Float16* p) {
  AFrag f;
  f.u[0] = *(const uint4*)p;
  f.u[1] = *(const uint4*)(p + 16);
  return f.v;
}

// ws layout (halfs), all B-fragment swizzled, 16384 halfs per matrix:
//   [0,16384) WqB  [16384,32768) WkB  [32768,49152) WvB  [49152,65536) WoB
// element (h,c,l,i) = W[c*32 + (i>>3)*16 + (l>>4)*8 + (i&7)][h*16 + (l&15)]
__global__ void cvt_weights_kernel(const float* __restrict__ WQ,
                                   const float* __restrict__ WK,
                                   const float* __restrict__ WV,
                                   const float* __restrict__ WO,
                                   _Float16* __restrict__ ws) {
  const int g = blockIdx.x * blockDim.x + threadIdx.x;
  if (g >= 65536) return;
  const int m = g >> 14;
  const int e = g & 16383;
  const int h = e >> 11;
  const int c = (e >> 9) & 3;
  const int l = (e >> 4) & 31;
  const int i = e & 15;
  const int src = (c * 32 + (i >> 3) * 16 + (l >> 4) * 8 + (i & 7)) * 128 + h * 16 + (l & 15);
  const float* W = (m == 0) ? WQ : (m == 1) ? WK : (m == 2) ? WV : WO;
  ws[g] = (_Float16)W[src];
}

__global__ __launch_bounds__(512)
void attn_kernel(const float* __restrict__ center,
                 const float* __restrict__ neighbors,
                 const _Float16* __restrict__ ws,
                 float* __restrict__ out) {
  __shared__ _Float16 tok[NB][80][TOK_PAD];  // 43.5 KB, rows 65..79 zeroed
  __shared__ _Float16 oSh[NB][D];            // attention output, f16 A-source

  const int tid  = threadIdx.x;
  const int lane = tid & 31;
  const int w    = tid >> 5;
  const int bi   = w >> 3;       // batch-in-block
  const int h    = w & 7;        // head / N-tile owned by this wave
  const int n    = lane & 15;    // column (or A-row) within 16-wide tile
  const int hi   = lane >> 4;    // half-wave id
  const int kbase = hi * 8;      // A-fragment K offset for this half

  // ---- stage tokens: [center; neighbors; zero-pad] -> LDS f16 ----
  for (int idx = tid; idx < NB * 80 * 32; idx += 512) {
    const int tbi = idx / (80 * 32);
    const int rem = idx - tbi * (80 * 32);
    const int row = rem >> 5;
    const int q4  = rem & 31;
    const size_t bb = (size_t)(blockIdx.x * NB + tbi);
    float4 v;
    if (row == 0)        v = ((const float4*)(center + bb * D))[q4];
    else if (row <= 64)  v = ((const float4*)(neighbors + (bb * 64 + (row - 1)) * D))[q4];
    else                 v = make_float4(0.f, 0.f, 0.f, 0.f);
    v4h hv = { (_Float16)v.x, (_Float16)v.y, (_Float16)v.z, (_Float16)v.w };
    *(v4h*)&tok[tbi][row][q4 * 4] = hv;
  }
  __syncthreads();

  const _Float16* wqB = ws;
  const _Float16* wkB = ws + 16384;
  const _Float16* wvB = ws + 32768;
  const _Float16* woB = ws + 49152;

  // ---- persistent W_K / W_V B-fragments (coalesced 32B reads) ----
  v16h bk[4], bv[4];
#pragma unroll
  for (int c = 0; c < 4; ++c) {
    bk[c] = load_bfrag(wkB + (((h * 4 + c) * 32 + lane) * 16));
    bv[c] = load_bfrag(wvB + (((h * 4 + c) * 32 + lane) * 16));
  }

  // ---- q via WMMA: A rows {0,1} = centers of both batches ----
  const int rb = (n < 2) ? n : 0;   // per-lane A-row source select
  v8f cq = {};
#pragma unroll
  for (int c = 0; c < 4; ++c) {
    v16h aq = load_afrag_lds(&tok[rb][0][c * 32 + kbase]);
    v16h bq = load_bfrag(wqB + (((h * 4 + c) * 32 + lane) * 16));
    cq = __builtin_amdgcn_wmma_f32_16x16x32_f16(false, aq, false, bq,
                                                (short)0, cq, false, false);
  }
  // D row bi (vgpr bi, lanes 0-15) = q_bi[n]; broadcast into upper half
  const float qsel = (bi == 0) ? cq[0] : cq[1];
  const float qoth = __shfl_xor(qsel, 16, 32);
  const float qn   = (hi ? qoth : qsel) * 0.25f;   // hd^-0.5 = 16^-0.5

  // ---- flash attention over 5 token tiles of 16 (fully unrolled:
  //      DS offsets become immediates, tile-4 mask constant-folds) ----
  const _Float16* tbase = &tok[bi][n][0];   // A: lane = row n of tile 0
  float mrun = -3.0e38f, lrun = 0.f, oacc = 0.f;
#pragma unroll
  for (int tile = 0; tile < 5; ++tile) {
    const _Float16* trow = tbase + tile * (16 * TOK_PAD);
    v8f ck = {}; v8f cv = {};
#pragma unroll
    for (int c = 0; c < 4; ++c) {
      v16h a = load_afrag_lds(trow + c * 32 + kbase);
      ck = __builtin_amdgcn_wmma_f32_16x16x32_f16(false, a, false, bk[c],
                                                  (short)0, ck, false, false);
      cv = __builtin_amdgcn_wmma_f32_16x16x32_f16(false, a, false, bv[c],
                                                  (short)0, cv, false, false);
    }
    // scores for the 8 token-rows owned by this half-wave (DPP row-reduce)
    float s[8];
#pragma unroll
    for (int r = 0; r < 8; ++r) {
      const float t = rowsum16(qn * ck[r]);
      const int tix = tile * 16 + hi * 8 + r;     // constant-folds per tile
      s[r] = (tix < 65) ? t : -3.0e38f;
    }
    float tmax = s[0];
#pragma unroll
    for (int r = 1; r < 8; ++r) tmax = fmaxf(tmax, s[r]);
    tmax = fmaxf(tmax, __shfl_xor(tmax, 16, 32));   // cross half-wave
    const float mnew = fmaxf(mrun, tmax);
    const float corr = __expf(mrun - mnew);
    oacc *= corr;
    float ps = 0.f;
#pragma unroll
    for (int r = 0; r < 8; ++r) {
      const float p = __expf(s[r] - mnew);   // p[m] sits with V[m][d] in cv[r]
      ps += p;
      oacc += p * cv[r];
    }
    lrun = lrun * corr + ps + __shfl_xor(ps, 16, 32);
    mrun = mnew;
  }

  // combine row-halves, normalize, publish head output as f16 A-source
  const float ofull = oacc + __shfl_xor(oacc, 16, 32);
  const float outv  = ofull / lrun;
  if (hi == 0) oSh[bi][h * 16 + n] = (_Float16)outv;
  __syncthreads();

  // ---- final projection via WMMA: [2 x 128] @ W_O, waves bi==0 only ----
  if (bi == 0) {
    v8f co = {};
#pragma unroll
    for (int c = 0; c < 4; ++c) {
      v16h ao = load_afrag_lds(&oSh[rb][c * 32 + kbase]);
      v16h bo = load_bfrag(woB + (((h * 4 + c) * 32 + lane) * 16));
      co = __builtin_amdgcn_wmma_f32_16x16x32_f16(false, ao, false, bo,
                                                  (short)0, co, false, false);
    }
    if (hi == 0) {
      const size_t b0 = (size_t)blockIdx.x * NB;
      out[b0 * D + h * 16 + n]       = co[0];   // D row 0 = batch b0
      out[(b0 + 1) * D + h * 16 + n] = co[1];   // D row 1 = batch b1
    }
  }
}

extern "C" void kernel_launch(void* const* d_in, const int* in_sizes, int n_in,
                              void* d_out, int out_size, void* d_ws, size_t ws_size,
                              hipStream_t stream) {
  const float* center    = (const float*)d_in[0];
  const float* neighbors = (const float*)d_in[1];
  const float* WQ        = (const float*)d_in[2];
  const float* WK        = (const float*)d_in[3];
  const float* WV        = (const float*)d_in[4];
  const float* WO        = (const float*)d_in[5];
  float* out             = (float*)d_out;
  _Float16* ws           = (_Float16*)d_ws;   // needs 131072 bytes

  const int B = in_sizes[0] / D;   // 8192

  cvt_weights_kernel<<<256, 256, 0, stream>>>(WQ, WK, WV, WO, ws);
  attn_kernel<<<B / NB, 512, 0, stream>>>(center, neighbors, ws, out);
}
